// Net_61478161874964
// MI455X (gfx1250) — matrix operand order repaired and checked by reading, not compiled, not run on hardware
//
#include <hip/hip_runtime.h>
#include <hip/hip_bf16.h>

// ---------------------------------------------------------------------------
// GNN pipeline for MI455X (gfx1250, wave32, WMMA).
//   gcn1: mean-aggregate(feat) @ W1^T + b1, ReLU   -> h1 [51200,100] bf16
//   gcn2: mean-aggregate(h1)   @ W2^T + b2, ReLU   -> h2 [51200,20]  f32
//   tail: graph mean-pool, 2x (single-token cross-attn + LayerNorm), MLP
// GEMMs run on v_wmma_f32_16x16x32_bf16 (K=100 padded to 128 -> 4 k-steps).
// ---------------------------------------------------------------------------

typedef __attribute__((ext_vector_type(16))) __bf16 v16bf;
typedef __attribute__((ext_vector_type(8)))  __bf16 v8bf;
typedef __attribute__((ext_vector_type(8)))  float  v8f;

#define N_NODES 51200
#define N_PER_G 100
#define NGRAPH  512
#define DEG     16
#define D_IN    100
#define H1DIM   100
#define DG      20
#define H1_STRIDE 112   // 7 tiles of 16 cols (cols 100..111 never read)

// ---- prep: W1 [100,100] -> bf16 [112][128] K-contiguous; W2 -> [32][128] ----
__global__ __launch_bounds__(256) void prep_kernel(const float* __restrict__ W1,
                                                   const float* __restrict__ W2,
                                                   __bf16* __restrict__ Wb1t,
                                                   __bf16* __restrict__ Wb2t) {
  int i = blockIdx.x * blockDim.x + threadIdx.x;
  if (i < 112 * 128) {
    int o = i >> 7, k = i & 127;
    float v = (o < 100 && k < 100) ? W1[o * 100 + k] : 0.0f;
    Wb1t[i] = (__bf16)v;
  }
  int j = i - 112 * 128;
  if (j >= 0 && j < 32 * 128) {
    int o = j >> 7, k = j & 127;
    float v = (o < 20 && k < 100) ? W2[o * 100 + k] : 0.0f;
    Wb2t[j] = (__bf16)v;
  }
}

// ---- GCN layer 1: one wave per 16-node tile -------------------------------
__global__ __launch_bounds__(32) void gcn1_kernel(const float* __restrict__ feat,
                                                  const int*   __restrict__ edge_src,
                                                  const float* __restrict__ b1,
                                                  const __bf16* __restrict__ Wb1t,
                                                  __bf16* __restrict__ h1) {
  __shared__ int eidx[16 * DEG];
  __shared__ __align__(32) __bf16 agg[16][128];
  const int tile = blockIdx.x;
  const int lane = threadIdx.x;

  for (int i = lane; i < 16 * DEG; i += 32) eidx[i] = edge_src[tile * 16 * DEG + i];
  __syncthreads();

  const float scale = 1.0f / (float)DEG;
  for (int m = 0; m < 16; ++m) {
    float a0 = 0.f, a1 = 0.f, a2 = 0.f, a3 = 0.f;
    #pragma unroll
    for (int e = 0; e < DEG; ++e) {
      const float* row = feat + eidx[m * DEG + e] * D_IN;
      a0 += row[lane];
      a1 += row[lane + 32];
      a2 += row[lane + 64];
      if (lane < 4) a3 += row[lane + 96];
    }
    agg[m][lane]      = (__bf16)(a0 * scale);
    agg[m][lane + 32] = (__bf16)(a1 * scale);
    agg[m][lane + 64] = (__bf16)(a2 * scale);
    agg[m][lane + 96] = (lane < 4) ? (__bf16)(a3 * scale) : (__bf16)0.0f; // zero-pad K 100..127
  }
  __syncthreads();

  // A fragments: lane holds row M = lane&15; K runs [c*32+hf*16, +16)
  const int mrow = lane & 15;
  const int hf   = lane >> 4;
  v16bf afrag[4];
  #pragma unroll
  for (int c = 0; c < 4; ++c) {
    int s = c * 32 + hf * 16;
    v8bf lo = *(const v8bf*)&agg[mrow][s];
    v8bf hi = *(const v8bf*)&agg[mrow][s + 8];
    afrag[c] = __builtin_shufflevector(lo, hi, 0,1,2,3,4,5,6,7,8,9,10,11,12,13,14,15);
  }

  const int n    = lane & 15;
  const int koff = hf * 16;
  #pragma unroll
  for (int t = 0; t < 7; ++t) {
    v8f acc = {};
    #pragma unroll
    for (int c = 0; c < 4; ++c) {
      v16bf b = *(const v16bf*)&Wb1t[(t * 16 + n) * 128 + c * 32 + koff];
      acc = __builtin_amdgcn_wmma_f32_16x16x32_bf16(false, afrag[c], false, b,
                                                    (short)0, acc, false, false);
    }
    const int col = t * 16 + n;
    if (col < 100) {
      float bias = b1[col];
      #pragma unroll
      for (int r = 0; r < 8; ++r) {
        float v = acc[r] + bias;
        v = v > 0.f ? v : 0.f;
        h1[(tile * 16 + r + hf * 8) * H1_STRIDE + col] = (__bf16)v;
      }
    }
  }
}

// ---- GCN layer 2: same structure, bf16 input, 2 N-tiles -------------------
__global__ __launch_bounds__(32) void gcn2_kernel(const __bf16* __restrict__ h1,
                                                  const int*   __restrict__ edge_src,
                                                  const float* __restrict__ b2,
                                                  const __bf16* __restrict__ Wb2t,
                                                  float* __restrict__ h2) {
  __shared__ int eidx[16 * DEG];
  __shared__ __align__(32) __bf16 agg[16][128];
  const int tile = blockIdx.x;
  const int lane = threadIdx.x;

  for (int i = lane; i < 16 * DEG; i += 32) eidx[i] = edge_src[tile * 16 * DEG + i];
  __syncthreads();

  const float scale = 1.0f / (float)DEG;
  for (int m = 0; m < 16; ++m) {
    float a0 = 0.f, a1 = 0.f, a2 = 0.f, a3 = 0.f;
    #pragma unroll
    for (int e = 0; e < DEG; ++e) {
      const __bf16* row = h1 + eidx[m * DEG + e] * H1_STRIDE;
      a0 += (float)row[lane];
      a1 += (float)row[lane + 32];
      a2 += (float)row[lane + 64];
      if (lane < 4) a3 += (float)row[lane + 96];
    }
    agg[m][lane]      = (__bf16)(a0 * scale);
    agg[m][lane + 32] = (__bf16)(a1 * scale);
    agg[m][lane + 64] = (__bf16)(a2 * scale);
    agg[m][lane + 96] = (lane < 4) ? (__bf16)(a3 * scale) : (__bf16)0.0f;
  }
  __syncthreads();

  const int mrow = lane & 15;
  const int hf   = lane >> 4;
  v16bf afrag[4];
  #pragma unroll
  for (int c = 0; c < 4; ++c) {
    int s = c * 32 + hf * 16;
    v8bf lo = *(const v8bf*)&agg[mrow][s];
    v8bf hi = *(const v8bf*)&agg[mrow][s + 8];
    afrag[c] = __builtin_shufflevector(lo, hi, 0,1,2,3,4,5,6,7,8,9,10,11,12,13,14,15);
  }

  const int n    = lane & 15;
  const int koff = hf * 16;
  #pragma unroll
  for (int t = 0; t < 2; ++t) {
    v8f acc = {};
    #pragma unroll
    for (int c = 0; c < 4; ++c) {
      v16bf b = *(const v16bf*)&Wb2t[(t * 16 + n) * 128 + c * 32 + koff];
      acc = __builtin_amdgcn_wmma_f32_16x16x32_bf16(false, afrag[c], false, b,
                                                    (short)0, acc, false, false);
    }
    const int col = t * 16 + n;
    if (col < DG) {
      float bias = b2[col];
      #pragma unroll
      for (int r = 0; r < 8; ++r) {
        float v = acc[r] + bias;
        v = v > 0.f ? v : 0.f;
        h2[(tile * 16 + r + hf * 8) * DG + col] = v;
      }
    }
  }
}

// ---- tail: pool + 2x (V@Wo cross-attn + LayerNorm) + MLP, one wave/graph --
__global__ __launch_bounds__(32) void tail_kernel(const float* __restrict__ h2,
    const float* __restrict__ self_feat, const float* __restrict__ x3d,
    const float* __restrict__ Wv2, const float* __restrict__ Wo2,
    const float* __restrict__ g2,  const float* __restrict__ be2,
    const float* __restrict__ Wv3, const float* __restrict__ Wo3,
    const float* __restrict__ g3,  const float* __restrict__ be3,
    const float* __restrict__ Wf1, const float* __restrict__ bf1,
    const float* __restrict__ Wf2, const float* __restrict__ bf2,
    float* __restrict__ out) {
  const int g = blockIdx.x;
  const int lane = threadIdx.x;
  __shared__ float vbuf[32];
  __shared__ float ybuf[20];
  __shared__ float fbuf[10];

  // graph mean-pool
  float hg = 0.f;
  if (lane < DG) {
    const float* base = h2 + g * N_PER_G * DG + lane;
    for (int p = 0; p < N_PER_G; ++p) hg += base[p * DG];
    hg *= 0.01f;
  }

  // attn2: V = self_feat @ Wv2^T (lane j owns v[j])
  {
    float s = 0.f;
    const float* x = self_feat + g * 200;
    const float* w = Wv2 + lane * 200;
    for (int p = 0; p < 200; ++p) s += x[p] * w[p];
    vbuf[lane] = s;
  }
  __syncthreads();
  if (lane < DG) {
    float z = 0.f;
    #pragma unroll
    for (int j = 0; j < 32; ++j) z += vbuf[j] * Wo2[lane * 32 + j];
    ybuf[lane] = hg + z;
  }
  __syncthreads();
  float hg1 = 0.f;
  {
    float mu = 0.f;
    for (int d = 0; d < DG; ++d) mu += ybuf[d];
    mu *= 0.05f;
    float var = 0.f;
    for (int d = 0; d < DG; ++d) { float t = ybuf[d] - mu; var += t * t; }
    var *= 0.05f;
    if (lane < DG)
      hg1 = (ybuf[lane] - mu) * rsqrtf(var + 1e-5f) * g2[lane] + be2[lane];
  }
  __syncthreads();

  // attn3
  {
    float s = 0.f;
    const float* x = x3d + g * 100;
    const float* w = Wv3 + lane * 100;
    for (int p = 0; p < 100; ++p) s += x[p] * w[p];
    vbuf[lane] = s;
  }
  __syncthreads();
  if (lane < DG) {
    float z = 0.f;
    #pragma unroll
    for (int j = 0; j < 32; ++j) z += vbuf[j] * Wo3[lane * 32 + j];
    ybuf[lane] = hg1 + z;
  }
  __syncthreads();
  float hg2 = 0.f;
  {
    float mu = 0.f;
    for (int d = 0; d < DG; ++d) mu += ybuf[d];
    mu *= 0.05f;
    float var = 0.f;
    for (int d = 0; d < DG; ++d) { float t = ybuf[d] - mu; var += t * t; }
    var *= 0.05f;
    if (lane < DG)
      hg2 = (ybuf[lane] - mu) * rsqrtf(var + 1e-5f) * g3[lane] + be3[lane];
  }
  __syncthreads();
  if (lane < DG) ybuf[lane] = hg2;
  __syncthreads();

  // MLP head
  if (lane < 10) {
    float f = bf1[lane];
    for (int d = 0; d < DG; ++d) f += ybuf[d] * Wf1[lane * DG + d];
    fbuf[lane] = f > 0.f ? f : 0.f;
  }
  __syncthreads();
  if (lane == 0) {
    float o = bf2[0];
    for (int i = 0; i < 10; ++i) o += fbuf[i] * Wf2[i];
    out[g] = o;
  }
}

extern "C" void kernel_launch(void* const* d_in, const int* in_sizes, int n_in,
                              void* d_out, int out_size, void* d_ws, size_t ws_size,
                              hipStream_t stream) {
  const float* feat      = (const float*)d_in[0];
  const int*   edge_src  = (const int*)  d_in[1];
  // d_in[2] = edge_dst: dst of edge e is e/DEG by construction; not needed
  const float* self_feat = (const float*)d_in[3];
  const float* x3d       = (const float*)d_in[4];
  const float* W1  = (const float*)d_in[5];
  const float* b1  = (const float*)d_in[6];
  const float* W2  = (const float*)d_in[7];
  const float* b2  = (const float*)d_in[8];
  // Wq2/Wk2 (9,10) unused: softmax over a length-1 token axis is identically 1
  const float* Wv2 = (const float*)d_in[11];
  const float* Wo2 = (const float*)d_in[12];
  const float* g2  = (const float*)d_in[13];
  const float* be2 = (const float*)d_in[14];
  // Wq3/Wk3 (15,16) unused
  const float* Wv3 = (const float*)d_in[17];
  const float* Wo3 = (const float*)d_in[18];
  const float* g3  = (const float*)d_in[19];
  const float* be3 = (const float*)d_in[20];
  const float* Wf1 = (const float*)d_in[21];
  const float* bf1 = (const float*)d_in[22];
  const float* Wf2 = (const float*)d_in[23];
  const float* bf2 = (const float*)d_in[24];

  // workspace layout (256B-aligned offsets)
  char* ws = (char*)d_ws;
  __bf16* Wb1t = (__bf16*)(ws);                       // 112*128*2 = 28672
  __bf16* Wb2t = (__bf16*)(ws + 28672);               //  32*128*2 =  8192
  __bf16* h1   = (__bf16*)(ws + 36864);               // 51200*112*2 = 11468800
  float*  h2   = (float*) (ws + 36864 + 11468800);    // 51200*20*4  =  4096000
  float*  outp = (float*)d_out;

  prep_kernel<<<(112 * 128 + 32 * 128 + 255) / 256, 256, 0, stream>>>(W1, W2, Wb1t, Wb2t);
  gcn1_kernel<<<N_NODES / 16, 32, 0, stream>>>(feat, edge_src, b1, Wb1t, h1);
  gcn2_kernel<<<N_NODES / 16, 32, 0, stream>>>(h1, edge_src, b2, Wb2t, h2);
  tail_kernel<<<NGRAPH, 32, 0, stream>>>(h2, self_feat, x3d, Wv2, Wo2, g2, be2,
                                         Wv3, Wo3, g3, be3, Wf1, bf1, Wf2, bf2, outp);
}
